// MultiheadAttention_88837103551298
// MI455X (gfx1250) — matrix operand order repaired and checked by reading, not compiled, not run on hardware
//
#include <hip/hip_runtime.h>

typedef __attribute__((ext_vector_type(16))) __bf16 v16bf;
typedef __attribute__((ext_vector_type(8)))  __bf16 v8bf;
typedef __attribute__((ext_vector_type(8)))  float  v8f;
typedef __attribute__((ext_vector_type(4)))  float  v4f;

#define L_DIM 2048
#define S_DIM 2048
#define N_DIM 2
#define E_DIM 1024
#define H_DIM 16
#define HD    64
#define NH    (N_DIM * H_DIM)   // 32
#define ROWS  (L_DIM * N_DIM)   // 4096

static __device__ __forceinline__ v8f wmma_bf16(v16bf a, v16bf b, v8f c) {
  // D = A(16x32 bf16) * B(32x16 bf16) + C(16x16 f32)
  return __builtin_amdgcn_wmma_f32_16x16x32_bf16(false, a, false, b, (short)0, c,
                                                 false, false);
}

static __device__ __forceinline__ v4f ld4(const float* p) {
  return *(const v4f*)__builtin_assume_aligned(p, 16);
}
static __device__ __forceinline__ v8bf ld8bf(const __bf16* p) {
  return *(const v8bf*)__builtin_assume_aligned(p, 16);
}
static __device__ __forceinline__ v8bf cvt8(v4f x, v4f y) {
  v8bf r;
  r[0] = (__bf16)x[0]; r[1] = (__bf16)x[1]; r[2] = (__bf16)x[2]; r[3] = (__bf16)x[3];
  r[4] = (__bf16)y[0]; r[5] = (__bf16)y[1]; r[6] = (__bf16)y[2]; r[7] = (__bf16)y[3];
  return r;
}
static __device__ __forceinline__ v16bf cat8(v8bf lo, v8bf hi) {
  return __builtin_shufflevector(lo, hi, 0, 1, 2, 3, 4, 5, 6, 7,
                                 8, 9, 10, 11, 12, 13, 14, 15);
}

// One async 16B copy global->LDS (GVS addressing, tracked by ASYNCcnt).
static __device__ __forceinline__ void async_b128(uint32_t lds, uint32_t voff,
                                                  const void* sbase) {
  asm volatile("global_load_async_to_lds_b128 %0, %1, %2 offset:0"
               :: "v"(lds), "v"(voff), "s"(sbase) : "memory");
}

// ---------------------------------------------------------------------------
// Kernel 1: fused QKV projection.  out(4096 x 3072) = X(4096x1024) @ W^T + b.
// Block = 64 rows x 64 cols: 4 waves share the async-staged B tile (each wave
// stages a quarter), each wave owns a 16-row A tile.  Double-buffered K-steps
// of 32 with ASYNCcnt waits + workgroup split barriers.  Epilogue scatters
// bf16: q/k into [head][t][d], v TRANSPOSED into [head][d][s].
// ---------------------------------------------------------------------------
__global__ __launch_bounds__(128) void proj_qkv(
    const float* __restrict__ query, const float* __restrict__ key,
    const float* __restrict__ value, const float* __restrict__ Win,
    const float* __restrict__ bin,
    __bf16* __restrict__ qh, __bf16* __restrict__ kh, __bf16* __restrict__ vt) {
  __shared__ float astage[4][2][16 * 36];   // per-wave A, pitch 36 floats
  __shared__ float bstage[2][64 * 36];      // shared B, pitch 36 floats

  const int wib  = threadIdx.x >> 5;
  const int lane = threadIdx.x & 31;
  const int hi   = lane >> 4;
  const int lr   = lane & 15;

  const int rb = blockIdx.x / 48;           // 0..63: 64-row block
  const int tc = blockIdx.x % 48;           // 0..47: 64-col chunk
  const int jt = tc / 16;                   // 0=q 1=k 2=v
  const float* X = (jt == 0) ? query : ((jt == 1) ? key : value);

  const int row0 = rb * 64 + wib * 16;      // wave's 16 rows
  const int j0   = tc * 64;                 // block's 64 cols
  const float* Abase = X + (size_t)row0 * E_DIM;     // wave-uniform
  const float* Bbase = Win + (size_t)j0 * E_DIM;     // block-uniform

  const uint32_t arl = (uint32_t)(lane >> 3), abc = (uint32_t)(lane & 7) * 16u;
  auto stageA = [&](int buf, uint32_t kbyte) {
    const uint32_t base = (uint32_t)(uintptr_t)&astage[wib][buf][0];
#pragma unroll
    for (int i = 0; i < 4; ++i) {
      const uint32_t row = arl + (uint32_t)i * 4u;
      async_b128(base + row * 144u + abc,
                 kbyte + row * (uint32_t)(E_DIM * 4) + abc, Abase);
    }
  };
  auto stageB = [&](int buf, uint32_t kbyte) {   // wave stages its 16 B rows
    const uint32_t base = (uint32_t)(uintptr_t)&bstage[buf][0];
    const uint32_t r0 = (uint32_t)(wib * 16) + arl;
#pragma unroll
    for (int i = 0; i < 4; ++i) {
      const uint32_t row = r0 + (uint32_t)i * 4u;
      async_b128(base + row * 144u + abc,
                 kbyte + row * (uint32_t)(E_DIM * 4) + abc, Bbase);
    }
  };

  v8f acc[4];
#pragma unroll
  for (int sub = 0; sub < 4; ++sub)
    acc[sub] = (v8f){0.f, 0.f, 0.f, 0.f, 0.f, 0.f, 0.f, 0.f};

  stageA(0, 0u);
  stageB(0, 0u);

  for (int kb = 0; kb < E_DIM; kb += 32) {
    const int b = (kb >> 5) & 1;
    if (kb + 32 < E_DIM) {
      asm volatile("s_wait_dscnt 0" ::: "memory");
      __syncthreads();                           // WAR on buffer b^1
      stageA(b ^ 1, (uint32_t)(kb + 32) * 4u);
      stageB(b ^ 1, (uint32_t)(kb + 32) * 4u);
      asm volatile("s_wait_asynccnt 8" ::: "memory");  // my buf-b copies done
    } else {
      asm volatile("s_wait_asynccnt 0" ::: "memory");
    }
    __syncthreads();                             // everyone's buf-b copies done

    const float* Al = &astage[wib][b][0];
    const float* Bl = &bstage[b][0];
    const float* ap = Al + lr * 36 + hi * 16;
    const v16bf a = cat8(cvt8(ld4(ap), ld4(ap + 4)),
                         cvt8(ld4(ap + 8), ld4(ap + 12)));
#pragma unroll
    for (int sub = 0; sub < 4; ++sub) {
      const float* bp = Bl + (size_t)(sub * 16 + lr) * 36 + hi * 16;
      const v16bf bf = cat8(cvt8(ld4(bp), ld4(bp + 4)),
                            cvt8(ld4(bp + 8), ld4(bp + 12)));
      acc[sub] = wmma_bf16(a, bf, acc[sub]);
    }
  }

#pragma unroll
  for (int sub = 0; sub < 4; ++sub) {
#pragma unroll
    for (int r = 0; r < 8; ++r) {
      const int m  = r + 8 * hi;
      const int gr = row0 + m;           // flattened (t*N + n)
      const int t  = gr >> 1;
      const int n  = gr & 1;
      const int j  = j0 + sub * 16 + lr;
      float val = acc[sub][r] + bin[j];
      if (jt == 0) val *= 0.125f;        // q scaling = 64^-0.5
      const int e  = j & (E_DIM - 1);
      const int h  = e >> 6, d = e & 63;
      const int hh = n * H_DIM + h;
      if (jt == 2) {                     // V stored transposed: [head][d][s]
        vt[((size_t)hh * HD + d) * S_DIM + t] = (__bf16)val;
      } else {
        __bf16* dst = (jt == 0) ? qh : kh;
        dst[(((size_t)hh * L_DIM + t) << 6) + d] = (__bf16)val;
      }
    }
  }
}

// ---------------------------------------------------------------------------
// Kernel 2: flash attention.  One wave per (head, 16-query-row block).
// K (row-major) and V^T tiles double-buffered into per-wave LDS slices via
// global_load_async_to_lds_b128 (ASYNCcnt); every fragment is a contiguous
// 16B vector load.  S-loop step 32: 2 score tiles (C init = bias tile),
// online softmax with half-wave shuffles, P repacked via LDS, 4 WMMAs P @ V.
// ---------------------------------------------------------------------------
__global__ __launch_bounds__(128) void flash_attn(
    const __bf16* __restrict__ qh, const __bf16* __restrict__ kh,
    const __bf16* __restrict__ vt, const float* __restrict__ bias,
    __bf16* __restrict__ aout) {
  __shared__ __bf16 kstage[4][2][32 * 64];   // 32 KB: K row-major [s][d]
  __shared__ __bf16 vstage[4][2][64 * 40];   // 40 KB: V^T [d][s], pitch 40
  __shared__ float  pbuf[4][16 * 32];        //  8 KB: P staging

  const int wib  = threadIdx.x >> 5;
  const int w    = blockIdx.x * 4 + wib;
  const int lane = threadIdx.x & 31;
  const int hi   = lane >> 4;
  const int lr   = lane & 15;

  const int nh = w >> 7;                 // head index 0..31
  const int qb = w & 127;                // query block 0..127
  const int l0 = qb * 16;
  float* P = pbuf[wib];

  const float*  bias_base = bias + ((size_t)nh * L_DIM + l0) * S_DIM;
  const __bf16* Kh = kh + (size_t)nh * S_DIM * HD;
  const __bf16* Vt = vt + (size_t)nh * HD * S_DIM;

  auto stageK = [&](int buf, uint32_t gbyte) {
    const uint32_t base = (uint32_t)(uintptr_t)&kstage[wib][buf][0] +
                          (uint32_t)lane * 16u;
    const uint32_t g = gbyte + (uint32_t)lane * 16u;
#pragma unroll
    for (int i = 0; i < 8; ++i)
      async_b128(base + (uint32_t)i * 512u, g + (uint32_t)i * 512u, Kh);
  };
  auto stageV = [&](int buf, uint32_t gbyte) {   // V^T 64x32, LDS pitch 80B
    const uint32_t base = (uint32_t)(uintptr_t)&vstage[wib][buf][0];
    const uint32_t dl = (uint32_t)(lane >> 2), bcol = (uint32_t)(lane & 3) * 16u;
#pragma unroll
    for (int i = 0; i < 8; ++i) {
      const uint32_t d = dl + (uint32_t)i * 8u;
      async_b128(base + d * 80u + bcol,
                 gbyte + d * (uint32_t)(S_DIM * 2) + bcol, Vt);
    }
  };

  // Q fragments: rows l0..l0+15, head dim 0..63 -> 2 A fragments, 16B loads
  const __bf16* Qb = qh + ((size_t)nh * L_DIM + l0) * HD + (size_t)lr * HD;
  v16bf qa[2];
#pragma unroll
  for (int c = 0; c < 2; ++c)
    qa[c] = cat8(ld8bf(Qb + c * 32 + hi * 8), ld8bf(Qb + c * 32 + 16 + hi * 8));

  v8f o0 = {0.f,0.f,0.f,0.f,0.f,0.f,0.f,0.f};
  v8f o1 = o0, o2 = o0, o3 = o0;
  float mrow[8], lrow[8];
#pragma unroll
  for (int r = 0; r < 8; ++r) { mrow[r] = -3.0e38f; lrow[r] = 0.f; }

  stageK(0, 0u);
  stageV(0, 0u);

  for (int s0 = 0; s0 < S_DIM; s0 += 32) {
    const int b = (s0 >> 5) & 1;

    if (s0 + 32 < S_DIM) {
      asm volatile("s_wait_dscnt 0" ::: "memory");   // WAR on buffer b^1
      stageK(b ^ 1, (uint32_t)(s0 + 32) * (HD * 2));
      stageV(b ^ 1, (uint32_t)(s0 + 32) * 2u);
      asm volatile("s_wait_asynccnt 16" ::: "memory");
    } else {
      asm volatile("s_wait_asynccnt 0" ::: "memory");
    }

    const __bf16* Kl  = &kstage[wib][b][0];
    const __bf16* Vtl = &vstage[wib][b][0];

    // ---- scores: two 16x16 tiles, C initialized with the bias tile --------
    v8f sc[2];
#pragma unroll
    for (int half = 0; half < 2; ++half) {
      const float* bp = bias_base + s0 + half * 16;
      v8f c;
#pragma unroll
      for (int r = 0; r < 8; ++r)
        c[r] = bp[(size_t)(r + 8 * hi) * S_DIM + lr];
      const __bf16* krow = Kl + (size_t)(half * 16 + lr) * HD;
#pragma unroll
      for (int c2 = 0; c2 < 2; ++c2) {
        const v16bf kf = cat8(ld8bf(krow + c2 * 32 + hi * 8),
                              ld8bf(krow + c2 * 32 + 16 + hi * 8));
        c = wmma_bf16(qa[c2], kf, c);
      }
      sc[half] = c;
    }

    if (s0 + 32 < S_DIM)   // pull the next bias tile toward the caches
      __builtin_prefetch(bias_base + (size_t)(8 * hi) * S_DIM + s0 + 32 + lr, 0, 1);

    // ---- online softmax; row m = r + 8*hi lives in one 16-lane half -------
#pragma unroll
    for (int r = 0; r < 8; ++r) {
      const float v0 = sc[0][r], v1 = sc[1][r];
      float mx = fmaxf(v0, v1);
#pragma unroll
      for (int msk = 1; msk <= 8; msk <<= 1)
        mx = fmaxf(mx, __shfl_xor(mx, msk, 32));
      const float mnew = fmaxf(mrow[r], mx);
      const float corr = __expf(mrow[r] - mnew);
      const float p0 = __expf(v0 - mnew);
      const float p1 = __expf(v1 - mnew);
      float ps = p0 + p1;
#pragma unroll
      for (int msk = 1; msk <= 8; msk <<= 1)
        ps += __shfl_xor(ps, msk, 32);
      lrow[r] = lrow[r] * corr + ps;
      mrow[r] = mnew;
      o0[r] *= corr; o1[r] *= corr; o2[r] *= corr; o3[r] *= corr;
      P[(r + 8 * hi) * 32 + lr]      = p0;   // P in plain 16x32 row-major
      P[(r + 8 * hi) * 32 + 16 + lr] = p1;
    }

    // ---- repack P as A fragment (16x32 bf16); same-wave LDS is in-order ---
    const float* pr = P + lr * 32 + hi * 8;
    const v16bf pa = cat8(cvt8(ld4(pr), ld4(pr + 4)),
                          cvt8(ld4(pr + 16), ld4(pr + 20)));

    // ---- O += P(16x32) @ V(32x64): 4 WMMAs, contiguous V^T loads ---------
#pragma unroll
    for (int j = 0; j < 4; ++j) {
      const __bf16* vp = Vtl + (size_t)(j * 16 + lr) * 40 + hi * 8;
      const v16bf vb = cat8(ld8bf(vp), ld8bf(vp + 16));
      v8f& o = (j == 0) ? o0 : ((j == 1) ? o1 : ((j == 2) ? o2 : o3));
      o = wmma_bf16(pa, vb, o);
    }
  }

  // ---- normalize and store bf16 into (t, n, e) layout ---------------------
  const int n = nh >> 4, h = nh & 15;
#pragma unroll
  for (int r = 0; r < 8; ++r) {
    const float inv = 1.0f / lrow[r];
    const int t = l0 + r + 8 * hi;
    __bf16* orow = aout + ((size_t)t * N_DIM + n) * E_DIM + h * HD + lr;
    orow[0]  = (__bf16)(o0[r] * inv);
    orow[16] = (__bf16)(o1[r] * inv);
    orow[32] = (__bf16)(o2[r] * inv);
    orow[48] = (__bf16)(o3[r] * inv);
  }
}

// ---------------------------------------------------------------------------
// Kernel 3: out = A(4096x1024 bf16) @ Wout^T + bout, f32 output (L,N,E).
// Same block-cooperative async-staged structure as kernel 1 (A is bf16).
// ---------------------------------------------------------------------------
__global__ __launch_bounds__(128) void out_proj(
    const __bf16* __restrict__ A, const float* __restrict__ Wout,
    const float* __restrict__ bout, float* __restrict__ out) {
  __shared__ __bf16 astage[4][2][16 * 40];  // per-wave A (bf16), pitch 40
  __shared__ float  bstage[2][64 * 36];     // shared B, pitch 36 floats

  const int wib  = threadIdx.x >> 5;
  const int lane = threadIdx.x & 31;
  const int hi   = lane >> 4;
  const int lr   = lane & 15;

  const int rb = blockIdx.x / 16;           // 0..63
  const int tc = blockIdx.x % 16;           // 0..15
  const int row0 = rb * 64 + wib * 16;
  const int j0   = tc * 64;
  const __bf16* Abase = A + (size_t)row0 * E_DIM;
  const float*  Bbase = Wout + (size_t)j0 * E_DIM;

  auto stageA = [&](int buf, uint32_t kbyte) {     // 16 rows x 64B
    const uint32_t base = (uint32_t)(uintptr_t)&astage[wib][buf][0];
    const uint32_t rl = (uint32_t)(lane >> 2), bcol = (uint32_t)(lane & 3) * 16u;
#pragma unroll
    for (int i = 0; i < 2; ++i) {
      const uint32_t row = rl + (uint32_t)i * 8u;
      async_b128(base + row * 80u + bcol,
                 kbyte + row * (uint32_t)(E_DIM * 2) + bcol, Abase);
    }
  };
  const uint32_t brl = (uint32_t)(lane >> 3), bbc = (uint32_t)(lane & 7) * 16u;
  auto stageB = [&](int buf, uint32_t kbyte) {     // wave's 16 B rows
    const uint32_t base = (uint32_t)(uintptr_t)&bstage[buf][0];
    const uint32_t r0 = (uint32_t)(wib * 16) + brl;
#pragma unroll
    for (int i = 0; i < 4; ++i) {
      const uint32_t row = r0 + (uint32_t)i * 4u;
      async_b128(base + row * 144u + bbc,
                 kbyte + row * (uint32_t)(E_DIM * 4) + bbc, Bbase);
    }
  };

  v8f acc[4];
#pragma unroll
  for (int sub = 0; sub < 4; ++sub)
    acc[sub] = (v8f){0.f, 0.f, 0.f, 0.f, 0.f, 0.f, 0.f, 0.f};

  stageA(0, 0u);
  stageB(0, 0u);

  for (int kb = 0; kb < E_DIM; kb += 32) {
    const int b = (kb >> 5) & 1;
    if (kb + 32 < E_DIM) {
      asm volatile("s_wait_dscnt 0" ::: "memory");
      __syncthreads();
      stageA(b ^ 1, (uint32_t)(kb + 32) * 2u);
      stageB(b ^ 1, (uint32_t)(kb + 32) * 4u);
      asm volatile("s_wait_asynccnt 6" ::: "memory");
    } else {
      asm volatile("s_wait_asynccnt 0" ::: "memory");
    }
    __syncthreads();

    const __bf16* Al = &astage[wib][b][0];
    const float*  Bl = &bstage[b][0];
    const __bf16* ap = Al + lr * 40 + hi * 16;
    const v16bf a = cat8(ld8bf(ap), ld8bf(ap + 8));
#pragma unroll
    for (int sub = 0; sub < 4; ++sub) {
      const float* bp = Bl + (size_t)(sub * 16 + lr) * 36 + hi * 16;
      const v16bf bf = cat8(cvt8(ld4(bp), ld4(bp + 4)),
                            cvt8(ld4(bp + 8), ld4(bp + 12)));
      acc[sub] = wmma_bf16(a, bf, acc[sub]);
    }
  }
#pragma unroll
  for (int sub = 0; sub < 4; ++sub) {
#pragma unroll
    for (int r = 0; r < 8; ++r) {
      const int m = r + 8 * hi;
      const int j = j0 + sub * 16 + lr;
      out[(size_t)(row0 + m) * E_DIM + j] = acc[sub][r] + bout[j];
    }
  }
}

// ---------------------------------------------------------------------------
extern "C" void kernel_launch(void* const* d_in, const int* in_sizes, int n_in,
                              void* d_out, int out_size, void* d_ws, size_t ws_size,
                              hipStream_t stream) {
  const float* query = (const float*)d_in[0];
  const float* key_  = (const float*)d_in[1];
  const float* value = (const float*)d_in[2];
  const float* bias  = (const float*)d_in[3];
  const float* Win   = (const float*)d_in[4];
  const float* bin   = (const float*)d_in[5];
  const float* Wout  = (const float*)d_in[6];
  const float* bout  = (const float*)d_in[7];

  const size_t HEAD_ELEMS = (size_t)NH * L_DIM * HD;   // 4 Mi bf16 each
  __bf16* qh   = (__bf16*)d_ws;
  __bf16* kh   = qh + HEAD_ELEMS;
  __bf16* vt   = kh + HEAD_ELEMS;      // transposed V: [head][d][s]
  __bf16* aout = vt + HEAD_ELEMS;

  // 64 row-blocks (64 rows) * 48 col-chunks (64 cols)
  proj_qkv<<<(ROWS / 64) * (3 * E_DIM / 64), 128, 0, stream>>>(
      query, key_, value, Win, bin, qh, kh, vt);
  // 32 heads * 128 query blocks, 4 waves/block
  flash_attn<<<(NH * (L_DIM / 16)) / 4, 128, 0, stream>>>(qh, kh, vt, bias, aout);
  // 64 row-blocks * 16 col-chunks
  out_proj<<<(ROWS / 64) * (E_DIM / 64), 128, 0, stream>>>(
      aout, Wout, bout, (float*)d_out);
}
